// Attention_2362232013200
// MI455X (gfx1250) — compile-verified, matrix operand order checked by reading, not run on hardware
//
#include <hip/hip_runtime.h>
#include <hip/hip_bf16.h>
#include <hip/hip_fp16.h>

// Problem constants (match reference)
#define BB 256
#define SS 196
#define DD 2048
#define HH 512

typedef __attribute__((ext_vector_type(2)))  float    v2f;
typedef __attribute__((ext_vector_type(8)))  float    v8f;
typedef __attribute__((ext_vector_type(16))) _Float16 v16h;

// ---------------------------------------------------------------------------
// Kernel 1: att_h[B,H] = h[B,D] @ W_h2att[H,D]^T + b_h2att   (fp32 WMMA)
// One wave (32 threads) per 16x16 output tile. grid = (H/16, B/16).
// ISA layout (CDNA5 ISA):
//   A 16x4 (MxK): lane holds row M=lane%16; VGPR0/1 = K{0,1} (lanes<16) or
//                 K{2,3} (lanes>=16) -> contiguous pair at k + 2*(lane>>4).
//   B 4x16 (KxN): mirrored, lane holds col N=lane%16, same K pairing.
//   C/D 16x16 f32: VGPR r -> M = r + 8*(lane>>4), N = lane%16.
// ---------------------------------------------------------------------------
__global__ __launch_bounds__(32) void h2att_wmma(
    const float* __restrict__ h,      // [B, D]
    const float* __restrict__ W,      // [H, D]
    const float* __restrict__ bias,   // [H]
    float* __restrict__ att_h)        // [B, H]
{
  const int lane   = threadIdx.x & 31;
  const int tile_n = blockIdx.x;            // along H
  const int tile_m = blockIdx.y;            // along B
  const int lo     = lane & 15;
  const int hi     = lane >> 4;

  const float* __restrict__ arow = h + (size_t)(tile_m * 16 + lo) * DD;
  const float* __restrict__ brow = W + (size_t)(tile_n * 16 + lo) * DD;

  v8f c = {};

#if __has_builtin(__builtin_amdgcn_wmma_f32_16x16x4_f32)
  const int kk = hi * 2;
  #pragma unroll 8
  for (int k = 0; k < DD; k += 4) {
    v2f a = *(const v2f*)(arow + k + kk);   // 8B aligned (k+kk even)
    v2f b = *(const v2f*)(brow + k + kk);
    c = __builtin_amdgcn_wmma_f32_16x16x4_f32(
        /*neg_a=*/false, a, /*neg_b=*/false, b,
        /*c_mod=*/(short)0, c, /*reuse_a=*/false, /*reuse_b=*/false);
  }
#else
  // Fallback: probe-confirmed f16 WMMA (K=32 per issue), on-the-fly convert.
  // 16-bit A 16x32 layout: VGPR v(0..3): K = hi*8 + 2v {,+1};
  //                        VGPR v(4..7): K = 16 + hi*8 + 2(v-4) {,+1}.
  for (int k = 0; k < DD; k += 32) {
    v16h a, b;
    #pragma unroll
    for (int v = 0; v < 8; ++v) {
      const int k0 = k + ((v & 4) ? 16 : 0) + hi * 8 + 2 * (v & 3);
      a[2 * v]     = (_Float16)arow[k0];
      a[2 * v + 1] = (_Float16)arow[k0 + 1];
      b[2 * v]     = (_Float16)brow[k0];
      b[2 * v + 1] = (_Float16)brow[k0 + 1];
    }
    c = __builtin_amdgcn_wmma_f32_16x16x32_f16(
        false, a, false, b, (short)0, c, false, false);
  }
#endif

  const int col = tile_n * 16 + lo;
  const float bv = bias[col];
  #pragma unroll
  for (int r = 0; r < 8; ++r) {
    const int row = tile_m * 16 + r + hi * 8;
    att_h[(size_t)row * HH + col] = c[r] + bv;
  }
}

// ---------------------------------------------------------------------------
// Kernel 2 (fused, bandwidth-critical): per batch row b
//   scores[s] = w_alpha . tanh(p_att_feats[b,s,:] + att_h[b,:]) + b_alpha
//   w = masked-renormalized softmax(scores)
//   out[b,:] = sum_s w[s] * att_feats[b,s,:]
// One 256-thread block (8 wave32) per b. float4 streaming for att_feats.
// ---------------------------------------------------------------------------
__global__ __launch_bounds__(256) void att_fused(
    const float* __restrict__ att_h,     // [B, H]
    const float* __restrict__ p_att,     // [B, S, H]
    const float* __restrict__ att_feats, // [B, S, D]
    const float* __restrict__ att_masks, // [B, S]
    const float* __restrict__ w_alpha,   // [H]
    const float* __restrict__ b_alpha,   // [1]
    float* __restrict__ out)             // [B, D]
{
  const int b    = blockIdx.x;
  const int t    = threadIdx.x;
  const int lane = t & 31;
  const int wave = t >> 5;

  __shared__ float sh_ah[HH];        // att_h row
  __shared__ float sh_sc[SS + 12];   // scores -> weights (padded)
  __shared__ float sh_red[8];        // per-wave reduction scratch

  // Stage att_h[b,:] (512 floats) in LDS.
  sh_ah[t]       = att_h[(size_t)b * HH + t];
  sh_ah[t + 256] = att_h[(size_t)b * HH + t + 256];
  __syncthreads();

  // ---- Phase 1: scores. Each wave owns slots s = wave, wave+8, ...
  const float* __restrict__ pbase = p_att + (size_t)b * SS * HH;
  for (int s = wave; s < SS; s += 8) {
    const float* __restrict__ pr = pbase + (size_t)s * HH;
    float acc = 0.f;
    #pragma unroll
    for (int j = 0; j < HH / 32; ++j) {     // 16 coalesced 128B wave reads
      const int idx = lane + 32 * j;
      acc += w_alpha[idx] * tanhf(pr[idx] + sh_ah[idx]);
    }
    #pragma unroll
    for (int off = 16; off > 0; off >>= 1)
      acc += __shfl_xor(acc, off, 32);
    if (lane == 0) sh_sc[s] = acc + b_alpha[0];
  }
  __syncthreads();

  // ---- Phase 2: masked softmax with renormalization.
  float v = (t < SS) ? sh_sc[t] : -3.4e38f;
  #pragma unroll
  for (int off = 16; off > 0; off >>= 1)
    v = fmaxf(v, __shfl_xor(v, off, 32));
  if (lane == 0) sh_red[wave] = v;
  __syncthreads();
  float mx = sh_red[0];
  #pragma unroll
  for (int i = 1; i < 8; ++i) mx = fmaxf(mx, sh_red[i]);
  __syncthreads();                           // done reading sh_red (max)

  float e = 0.f;
  if (t < SS)
    e = __expf(sh_sc[t] - mx) * att_masks[(size_t)b * SS + t];
  float sum = e;
  #pragma unroll
  for (int off = 16; off > 0; off >>= 1)
    sum += __shfl_xor(sum, off, 32);
  if (lane == 0) sh_red[wave] = sum;
  __syncthreads();
  float tot = sh_red[0];
  #pragma unroll
  for (int i = 1; i < 8; ++i) tot += sh_red[i];
  if (t < SS) sh_sc[t] = e / tot;            // final weights in LDS
  __syncthreads();

  // ---- Phase 3: out[b,:] = sum_s w[s] * att_feats[b,s,:]  (HBM-bound)
  // Thread t owns float4 columns t and t+256 (2048 floats / row).
  const float4* __restrict__ feats =
      (const float4*)(att_feats + (size_t)b * SS * DD);
  float4 acc0 = make_float4(0.f, 0.f, 0.f, 0.f);
  float4 acc1 = make_float4(0.f, 0.f, 0.f, 0.f);
  #pragma unroll 2
  for (int s = 0; s < SS; ++s) {
    const float w = sh_sc[s];
    const float4* __restrict__ row = feats + (size_t)s * (DD / 4);
    if (s + 4 < SS) {  // keep HBM pipe full (global_prefetch_b8)
      __builtin_prefetch((const void*)(row + 4 * (DD / 4) + t), 0, 1);
      __builtin_prefetch((const void*)(row + 4 * (DD / 4) + t + 256), 0, 1);
    }
    const float4 x0 = row[t];
    const float4 x1 = row[t + 256];
    acc0.x += w * x0.x; acc0.y += w * x0.y; acc0.z += w * x0.z; acc0.w += w * x0.w;
    acc1.x += w * x1.x; acc1.y += w * x1.y; acc1.z += w * x1.z; acc1.w += w * x1.w;
  }
  float4* __restrict__ o = (float4*)(out + (size_t)b * DD);
  o[t]       = acc0;
  o[t + 256] = acc1;
}

// ---------------------------------------------------------------------------
extern "C" void kernel_launch(void* const* d_in, const int* in_sizes, int n_in,
                              void* d_out, int out_size, void* d_ws, size_t ws_size,
                              hipStream_t stream) {
  const float* h           = (const float*)d_in[0];  // [B,D]
  const float* att_feats   = (const float*)d_in[1];  // [B,S,D]
  const float* p_att_feats = (const float*)d_in[2];  // [B,S,H]
  const float* att_masks   = (const float*)d_in[3];  // [B,S]
  const float* W_h2att     = (const float*)d_in[4];  // [H,D]
  const float* b_h2att     = (const float*)d_in[5];  // [H]
  const float* w_alpha     = (const float*)d_in[6];  // [H]
  const float* b_alpha     = (const float*)d_in[7];  // [1]

  float* att_h = (float*)d_ws;                       // B*H fp32 = 512 KB

  dim3 g1(HH / 16, BB / 16);                         // 32 x 16 = 512 tiles
  h2att_wmma<<<g1, 32, 0, stream>>>(h, W_h2att, b_h2att, att_h);

  att_fused<<<BB, 256, 0, stream>>>(att_h, p_att_feats, att_feats, att_masks,
                                    w_alpha, b_alpha, (float*)d_out);
}
